// LSTMAE_20306605375642
// MI455X (gfx1250) — matrix operand order repaired and checked by reading, not compiled, not run on hardware
//
#include <hip/hip_runtime.h>
#include <hip/hip_bf16.h>

// ---------------------------------------------------------------------------
// LSTM autoencoder forward (eval mode), CDNA5 / gfx1250, wave32 + WMMA.
//
// B=1024, T=512, F=64, H=128, 4H=512, K=F+H=192.
// 64 workgroups x 256 threads (8 waves). Each workgroup owns a 16-row batch
// tile and runs the full sequential recurrence with f16 weights in LDS and
// v_wmma_f32_16x16x32_f16 for every matmul. Cell state c lives in WMMA
// accumulator-layout registers per wave (wave w owns hidden cols w*16..+15).
// Gate biases live in persistent v8f fragments and enter the GEMM as the C
// operand of the first K-chunk WMMA (no per-step broadcast movs).
// ---------------------------------------------------------------------------

typedef _Float16 v16h __attribute__((ext_vector_type(16)));
typedef _Float16 v8h  __attribute__((ext_vector_type(8)));
typedef float    v8f  __attribute__((ext_vector_type(8)));

#define BB 1024
#define TT 512
#define FF 64
#define HH 128
#define G4 512    // 4*H
#define KC 192    // F + H (concat input|hidden K dimension)

// LDS layout (bytes):
//  sW    : [512][192] f16  gate weights, W^T (row = output col, K contiguous)
//  sWo   : [ 64][128] f16  W_out^T (decoder projection)
//  sX    : [ 16][192] f16  Xcat = [x_t (64) | h (128)] per batch row
//  sOut  : [ 16][ 64] f32  decoder projection output
//  sB    : [512]      f32  b_ih + b_hh
//  sBo   : [ 64]      f32  b_out
//  sRed  : [256]      f32  rec_err reduction
#define OFF_W    0
#define OFF_WO   (OFF_W   + G4*KC*2)        // 196608
#define OFF_X    (OFF_WO  + FF*HH*2)        // 212992
#define OFF_OUT  (OFF_X   + 16*KC*2)        // 219136
#define OFF_B    (OFF_OUT + 16*FF*4)        // 223232
#define OFF_BO   (OFF_B   + G4*4)           // 225280
#define OFF_RED  (OFF_BO  + FF*4)           // 225536
#define SMEM_BYTES (OFF_RED + 256*4)        // 226560  (<= 320KB WGP LDS)

__device__ __forceinline__ float fast_sig(float x) {
    return 1.0f / (1.0f + __expf(-x));
}
__device__ __forceinline__ float fast_tanh(float x) {
    float e = __expf(-2.0f * fabsf(x));
    float r = (1.0f - e) / (1.0f + e);
    return (x >= 0.0f) ? r : -r;
}

// Cooperative load: gate weights -> f16 W^T in LDS, bias sum -> f32 LDS.
__device__ __forceinline__ void load_gate_weights(
    _Float16* __restrict__ sW, float* __restrict__ sB,
    const float* __restrict__ Wih, const float* __restrict__ Whh,
    const float* __restrict__ bih, const float* __restrict__ bhh, int tid)
{
    for (int i = tid; i < G4 * KC; i += 256) {
        int col = i / KC;
        int k   = i - col * KC;
        float w = (k < FF) ? Wih[col * FF + k] : Whh[col * HH + (k - FF)];
        sW[i] = (_Float16)w;
    }
    for (int c = tid; c < G4; c += 256) sB[c] = bih[c] + bhh[c];
}

// Build per-wave bias fragments (C-operand layout: splat over the 8 rows this
// lane covers; column = hcol + (lane&15)). Done once per phase.
__device__ __forceinline__ void build_bias_frags(
    const float* __restrict__ sB, int wave, int lane, v8f bias[4])
{
    const int nlo = lane & 15;
    const int hcol = wave * 16;
    #pragma unroll
    for (int g = 0; g < 4; ++g) {
        float b = sB[g * HH + hcol + nlo];
        #pragma unroll
        for (int e = 0; e < 8; ++e) bias[g][e] = b;
    }
}

// A-fragment loader: contiguous k order (B loads match). Lanes<16 hold K{0..15},
// lanes>=16 hold K{16..31} of the 32-wide chunk starting at kbase.
__device__ __forceinline__ v16h load_a_frag(
    const _Float16* __restrict__ arow, int kbase, bool hi)
{
    const int k0 = kbase + (hi ? 16 : 0);
    v8h alo = *(const v8h*)(arow + k0);
    v8h ahi = *(const v8h*)(arow + k0 + 8);
    return __builtin_shufflevector(alo, ahi,
               0,1,2,3,4,5,6,7,8,9,10,11,12,13,14,15);
}

// One LSTM cell step for this workgroup's 16 batch rows.
// Wave w computes gate tiles {i,f,g,o} for hidden cols [w*16, w*16+16) with
// 24 v_wmma_f32_16x16x32_f16 (bias enters as C of the first chunk), applies
// activations in accumulator layout, updates its resident c fragment, and
// writes h (f16) back into sX.
__device__ __forceinline__ void lstm_gates(
    const _Float16* __restrict__ sW, const v8f bias[4],
    _Float16* __restrict__ sX, int wave, int lane, v8f& cfrag)
{
    const int  nlo = lane & 15;
    const bool hi  = (lane >= 16);
    const int  hcol = wave * 16;

    const _Float16* arow = sX + nlo * KC;      // lane L and L+16: batch row L&15
    const int boff = (hi ? 16 : 0);            // B: lanes>=16 carry K 16..31

    v8f acc[4];
    {   // kb = 0: accumulate on top of the persistent bias fragments
        v16h a = load_a_frag(arow, 0, hi);
        #pragma unroll
        for (int g = 0; g < 4; ++g) {
            const _Float16* brow = sW + (size_t)(g * HH + hcol + nlo) * KC + boff;
            v16h b = *(const v16h*)brow;
            acc[g] = __builtin_amdgcn_wmma_f32_16x16x32_f16(
                         false, a, false, b, (short)0, bias[g], false, false);
        }
    }
    #pragma unroll
    for (int kb = 1; kb < 6; ++kb) {
        v16h a = load_a_frag(arow, kb * 32, hi);
        #pragma unroll
        for (int g = 0; g < 4; ++g) {
            const _Float16* brow =
                sW + (size_t)(g * HH + hcol + nlo) * KC + kb * 32 + boff;
            v16h b = *(const v16h*)brow;
            acc[g] = __builtin_amdgcn_wmma_f32_16x16x32_f16(
                         false, a, false, b, (short)0, acc[g], false, false);
        }
    }

    // Activations + state update in C/D fragment layout:
    // element e -> row M = e + (hi?8:0), col N = hcol + (lane&15).
    _Float16* hdst = sX + FF + hcol + nlo;
    #pragma unroll
    for (int e = 0; e < 8; ++e) {
        float ig = fast_sig (acc[0][e]);
        float fg = fast_sig (acc[1][e]);
        float gg = fast_tanh(acc[2][e]);
        float og = fast_sig (acc[3][e]);
        float c  = fg * cfrag[e] + ig * gg;
        cfrag[e] = c;
        float h  = og * fast_tanh(c);
        int m = e + (hi ? 8 : 0);
        hdst[m * KC] = (_Float16)h;
    }
}

__global__ __launch_bounds__(256)
void lstm_ae_kernel(
    const float* __restrict__ ts,        // [B,T,F]
    const float* __restrict__ Wih_enc,   // [4H,F]
    const float* __restrict__ Whh_enc,   // [4H,H]
    const float* __restrict__ bih_enc,   // [4H]
    const float* __restrict__ bhh_enc,   // [4H]
    const float* __restrict__ Wih_dec,   // [4H,F]
    const float* __restrict__ Whh_dec,   // [4H,H]
    const float* __restrict__ bih_dec,   // [4H]
    const float* __restrict__ bhh_dec,   // [4H]
    const float* __restrict__ Wout,      // [F,H]
    const float* __restrict__ bout,      // [F]
    float* __restrict__ out)             // c_enc[B,H] | rec_err[B] | last[B,F]
{
    extern __shared__ char smem[];
    _Float16* sW   = (_Float16*)(smem + OFF_W);
    _Float16* sWo  = (_Float16*)(smem + OFF_WO);
    _Float16* sX   = (_Float16*)(smem + OFF_X);
    float*    sOut = (float*)   (smem + OFF_OUT);
    float*    sB   = (float*)   (smem + OFF_B);
    float*    sBo  = (float*)   (smem + OFF_BO);
    float*    sRed = (float*)   (smem + OFF_RED);

    const int tid  = threadIdx.x;
    const int lane = tid & 31;
    const int wave = tid >> 5;
    const int nlo  = lane & 15;
    const bool hi  = (lane >= 16);
    const int r0   = blockIdx.x * 16;          // batch tile base row

    // ---- init: encoder weights, h=c=0, bias fragments ---------------------
    load_gate_weights(sW, sB, Wih_enc, Whh_enc, bih_enc, bhh_enc, tid);
    for (int i = tid; i < 16 * KC; i += 256) sX[i] = (_Float16)0.0f;
    __syncthreads();
    v8f bias[4];
    build_bias_frags(sB, wave, lane, bias);
    v8f cfrag = {};   // resident cell state fragment (16 rows x 16 hidden)

    // ---- encoder: 512 steps ----------------------------------------------
    for (int t = 0; t < TT; ++t) {
        __syncthreads();                       // prior h stores done
        {   // stage x_t (f32 global -> f16 LDS), 4 elems/thread
            int row = tid >> 4, c0 = (tid & 15) * 4;
            const float4 x4 = *(const float4*)
                (ts + ((size_t)(r0 + row) * TT + t) * FF + c0);
            _Float16* xd = sX + row * KC + c0;
            xd[0] = (_Float16)x4.x; xd[1] = (_Float16)x4.y;
            xd[2] = (_Float16)x4.z; xd[3] = (_Float16)x4.w;
        }
        __syncthreads();                       // x_t visible
        lstm_gates(sW, bias, sX, wave, lane, cfrag);
    }

    // ---- phase switch: emit c_enc, swap in decoder weights ----------------
    #pragma unroll
    for (int e = 0; e < 8; ++e) {
        int m = e + (hi ? 8 : 0);
        out[(size_t)(r0 + m) * HH + wave * 16 + nlo] = cfrag[e];
    }
    __syncthreads();                           // all gate reads of sW done
    load_gate_weights(sW, sB, Wih_dec, Whh_dec, bih_dec, bhh_dec, tid);
    for (int i = tid; i < FF * HH; i += 256) sWo[i] = (_Float16)Wout[i];
    for (int c = tid; c < FF; c += 256) sBo[c] = bout[c];
    __syncthreads();                           // decoder weights/biases visible
    build_bias_frags(sB, wave, lane, bias);
    v8f pbias;                                 // projection bias fragment
    {
        float b = (wave < 4) ? sBo[wave * 16 + nlo] : 0.0f;
        #pragma unroll
        for (int e = 0; e < 8; ++e) pbias[e] = b;
    }

    // ---- decoder: 512 autoregressive steps --------------------------------
    float acc_rec = 0.0f;
    for (int k = 0; k < TT; ++k) {
        __syncthreads();                       // h stores visible
        if (wave < 4) {                        // projection: out_i = h @ Wout^T + b
            const int cb = wave * 16;
            const _Float16* arow = sX + nlo * KC;
            const int boff = (hi ? 16 : 0);
            v8f po;
            {   // kb = 0 rides on the persistent projection-bias fragment
                v16h a = load_a_frag(arow, FF, hi);
                const _Float16* brow = sWo + (size_t)(cb + nlo) * HH + boff;
                v16h b = *(const v16h*)brow;
                po = __builtin_amdgcn_wmma_f32_16x16x32_f16(
                         false, a, false, b, (short)0, pbias, false, false);
            }
            #pragma unroll
            for (int kb = 1; kb < 4; ++kb) {   // K = 128 (h region at K off 64)
                v16h a = load_a_frag(arow, FF + kb * 32, hi);
                const _Float16* brow =
                    sWo + (size_t)(cb + nlo) * HH + kb * 32 + boff;
                v16h b = *(const v16h*)brow;
                po = __builtin_amdgcn_wmma_f32_16x16x32_f16(
                         false, a, false, b, (short)0, po, false, false);
            }
            // scatter: f32 for rec_err, f16 into Xcat x-region for the cell
            #pragma unroll
            for (int e = 0; e < 8; ++e) {
                int m = e + (hi ? 8 : 0);
                int col = cb + nlo;
                sOut[m * FF + col] = po[e];
                sX  [m * KC + col] = (_Float16)po[e];
            }
        }
        __syncthreads();                       // out_i visible
        {   // rec_err vs ts[:, T-1-k, :]; k==0 is output[:, -1, :]
            int row = tid >> 4, c0 = (tid & 15) * 4;
            const float* o4 = sOut + row * FF + c0;
            const int tsi = TT - 1 - k;
            const float4 t4 = *(const float4*)
                (ts + ((size_t)(r0 + row) * TT + tsi) * FF + c0);
            acc_rec += fabsf(o4[0]*o4[0] - t4.x*t4.x)
                     + fabsf(o4[1]*o4[1] - t4.y*t4.y)
                     + fabsf(o4[2]*o4[2] - t4.z*t4.z)
                     + fabsf(o4[3]*o4[3] - t4.w*t4.w);
            if (k == 0) {
                float4 ov = make_float4(o4[0], o4[1], o4[2], o4[3]);
                *(float4*)(out + (size_t)BB*HH + BB
                               + (size_t)(r0 + row) * FF + c0) = ov;
            }
        }
        lstm_gates(sW, bias, sX, wave, lane, cfrag);
    }

    // ---- rec_err reduction (16 threads per batch row) ---------------------
    __syncthreads();
    sRed[tid] = acc_rec;
    __syncthreads();
    if ((tid & 15) == 0) {
        float s = 0.0f;
        const int base = tid & ~15;
        #pragma unroll
        for (int j = 0; j < 16; ++j) s += sRed[base + j];
        out[(size_t)BB * HH + (r0 + (tid >> 4))] = s;
    }
}

extern "C" void kernel_launch(void* const* d_in, const int* in_sizes, int n_in,
                              void* d_out, int out_size, void* d_ws, size_t ws_size,
                              hipStream_t stream) {
    (void)in_sizes; (void)n_in; (void)out_size; (void)d_ws; (void)ws_size;
    const float* ts      = (const float*)d_in[0];
    const float* Wih_enc = (const float*)d_in[1];
    const float* Whh_enc = (const float*)d_in[2];
    const float* bih_enc = (const float*)d_in[3];
    const float* bhh_enc = (const float*)d_in[4];
    const float* Wih_dec = (const float*)d_in[5];
    const float* Whh_dec = (const float*)d_in[6];
    const float* bih_dec = (const float*)d_in[7];
    const float* bhh_dec = (const float*)d_in[8];
    const float* Wout    = (const float*)d_in[9];
    const float* bout    = (const float*)d_in[10];
    float* out = (float*)d_out;

    dim3 grid(BB / 16);   // 64 workgroups, one 16-row batch tile each
    dim3 block(256);      // 8 wave32 waves
    lstm_ae_kernel<<<grid, block, SMEM_BYTES, stream>>>(
        ts, Wih_enc, Whh_enc, bih_enc, bhh_enc,
        Wih_dec, Whh_dec, bih_dec, bhh_dec, Wout, bout, out);
}
